// DGCN_HGN_9543417332135
// MI455X (gfx1250) — matrix operand order repaired and checked by reading, not compiled, not run on hardware
//
#include <hip/hip_runtime.h>
#include <math.h>

// ---------------------------------------------------------------------------
// CDNA5 (gfx1250) wave32 WMMA types
// ---------------------------------------------------------------------------
typedef __attribute__((ext_vector_type(16))) __bf16 v16bf;
typedef __attribute__((ext_vector_type(8)))  __bf16 v8bf;
typedef __attribute__((ext_vector_type(8)))  float  v8f;
typedef __attribute__((ext_vector_type(4)))  unsigned int v4u;
typedef __attribute__((ext_vector_type(8)))  int v8i;
typedef __attribute__((ext_vector_type(4)))  int v4i;

#if defined(__has_builtin)
#  if __has_builtin(__builtin_amdgcn_tensor_load_to_lds) && __has_builtin(__builtin_amdgcn_s_wait_tensorcnt)
#    define USE_TDM 1
#  endif
#endif
#ifndef USE_TDM
#  define USE_TDM 0
#endif

__device__ __forceinline__ __bf16 f2bf(float f) {
    union { float f; unsigned u; } x; x.f = f;
    unsigned r = x.u + 0x7fffu + ((x.u >> 16) & 1u);   // round-to-nearest-even
    unsigned short h = (unsigned short)(r >> 16);
    __bf16 o; __builtin_memcpy(&o, &h, 2);
    return o;
}
__device__ __forceinline__ float bf2f(__bf16 b) {
    unsigned short h; __builtin_memcpy(&h, &b, 2);
    union { unsigned u; float f; } x; x.u = ((unsigned)h) << 16;
    return x.f;
}
__device__ __forceinline__ float fastsig(float x) {
    return __builtin_amdgcn_rcpf(1.0f + __expf(-x));
}
__device__ __forceinline__ float fasttanh(float x) {
    return 1.0f - 2.0f * __builtin_amdgcn_rcpf(__expf(2.0f * x) + 1.0f);
}
// fragment = two contiguous 16B chunks in LDS
__device__ __forceinline__ v16bf frag_cat(const __bf16* lo, const __bf16* hi) {
    v8bf a = *(const v8bf*)lo;
    v8bf b = *(const v8bf*)hi;
    return __builtin_shufflevector(a, b, 0,1,2,3,4,5,6,7,8,9,10,11,12,13,14,15);
}

// ---------------------------------------------------------------------------
// LDS staging helpers (128 threads). As: 64 rows x 32 k. Bs: transposed, [n][k].
// Row stride 40 bf16 = 80 B -> every 8-element boundary is 16B aligned and
// rows hit distinct bank groups (gcd(20,64)=4 -> 16-row cycle).
// ---------------------------------------------------------------------------
__device__ __forceinline__ void stA_valu(const float* Ab, int row0, int K, int kk,
                                         int tid, __bf16 (*As)[40]) {
    #pragma unroll
    for (int i = 0; i < 4; ++i) {                    // 512 float4 = 64x32
        int idx = i * 128 + tid;
        int r = idx >> 3, c4 = idx & 7;
        const float4 q = *(const float4*)(Ab + (size_t)(row0 + r) * K + kk + c4 * 4);
        union { __bf16 h[4]; uint2 u; } p;
        p.h[0] = f2bf(q.x); p.h[1] = f2bf(q.y); p.h[2] = f2bf(q.z); p.h[3] = f2bf(q.w);
        *(uint2*)&As[r][c4 * 4] = p.u;
    }
}
__device__ __forceinline__ void stA_valu(const __bf16* Ab, int row0, int K, int kk,
                                         int tid, __bf16 (*As)[40]) {
    #pragma unroll
    for (int i = 0; i < 2; ++i) {                    // 256 x (8 bf16)
        int idx = i * 128 + tid;
        int r = idx >> 2, c8 = idx & 3;
        *(uint4*)&As[r][c8 * 8] = *(const uint4*)(Ab + (size_t)(row0 + r) * K + kk + c8 * 8);
    }
}

#if USE_TDM
// Tensor Data Mover: DMA one 32(k) x 64(rows) bf16 tile, global->LDS, with
// hardware row padding 64B data + 16B pad = 80B LDS row stride. Issued by one
// wave per block; tracked by TENSORcnt. Descriptor per ISA 08_async_tensor 8.3/8.4.
__device__ __forceinline__ void stA_tdm(const __bf16* Ab, int row0, int K, int kk,
                                        __bf16 (*As)[40]) {
    unsigned long long g = (unsigned long long)(const void*)(Ab + (size_t)row0 * K + kk);
    unsigned lds = (unsigned)(unsigned long long)(void*)&As[0][0];  // addrspace(3) offset in low 32b
    v4u g0; v8i g1; v4i g2, g3; v8i g4;
    g0[0] = 1u;                                                     // count=1, is_restore=0, gather off
    g0[1] = lds;                                                    // lds_addr (bytes)
    g0[2] = (unsigned)g;                                            // global_addr[31:0]
    g0[3] = (unsigned)((g >> 32) & 0x01FFFFFFull) | (2u << 30);     // global_addr[56:32] | type=2
    g1[0] = (int)((1u << 16) | (1u << 20) | (3u << 22) | (3u << 25)); // data_size=2B, pad_en, ivl=16dw, amt=4dw
    g1[1] = (int)(32u << 16);   // atomic_barrier=0 | tensor_dim0[15:0]=32
    g1[2] = (int)(64u << 16);   // tensor_dim0[31:16]=0 | tensor_dim1[15:0]=64
    g1[3] = (int)(32u << 16);   // tensor_dim1[31:16]=0 | tile_dim0=32
    g1[4] = 64;                 // tile_dim1=64 | tile_dim2=0
    g1[5] = K;                  // tensor_dim0_stride[31:0] (elements)
    g1[6] = 0;                  // stride[47:32]=0 | tensor_dim1_stride lo=0
    g1[7] = 0;
    g2[0] = g2[1] = g2[2] = g2[3] = 0;
    g3[0] = g3[1] = g3[2] = g3[3] = 0;
    g4[0] = g4[1] = g4[2] = g4[3] = 0;
    g4[4] = g4[5] = g4[6] = g4[7] = 0;
    __builtin_amdgcn_tensor_load_to_lds(g0, g1, g2, g3, g4, 0);
}
#endif

__device__ __forceinline__ void stageA(const float* Ab, int row0, int K, int kk,
                                       int tid, __bf16 (*As)[40]) {
    stA_valu(Ab, row0, K, kk, tid, As);
}
__device__ __forceinline__ void stageA(const __bf16* Ab, int row0, int K, int kk,
                                       int tid, __bf16 (*As)[40]) {
#if USE_TDM
    if (tid < 32) stA_tdm(Ab, row0, K, kk, As);
#else
    stA_valu(Ab, row0, K, kk, tid, As);
#endif
}
// wait (issuing wave only) for outstanding TDM before the block barrier
__device__ __forceinline__ void waitA(const float*, int) {}
__device__ __forceinline__ void waitA(const __bf16*, int tid) {
#if USE_TDM
    if (tid < 32) __builtin_amdgcn_s_wait_tensorcnt(0);
#endif
}

// B row-major KxN -> Bs[n][k] (transpose in LDS)
__device__ __forceinline__ void stBn(const float* Bb, int col0, int ldb, int kk,
                                     int tid, __bf16 (*Bs)[40]) {
    #pragma unroll
    for (int i = 0; i < 4; ++i) {                    // 512 float4 = 32x64
        int idx = i * 128 + tid;
        int k = idx >> 4, n4 = idx & 15;
        const float4 q = *(const float4*)(Bb + (size_t)(kk + k) * ldb + col0 + n4 * 4);
        Bs[n4 * 4 + 0][k] = f2bf(q.x); Bs[n4 * 4 + 1][k] = f2bf(q.y);
        Bs[n4 * 4 + 2][k] = f2bf(q.z); Bs[n4 * 4 + 3][k] = f2bf(q.w);
    }
}
__device__ __forceinline__ void stBn(const __bf16* Bb, int col0, int ldb, int kk,
                                     int tid, __bf16 (*Bs)[40]) {
    #pragma unroll
    for (int i = 0; i < 8; ++i) {                    // 1024 dwords = 32x64
        int idx = i * 128 + tid;
        int k = idx >> 5, n2 = idx & 31;
        unsigned int u = *(const unsigned int*)(Bb + (size_t)(kk + k) * ldb + col0 + n2 * 2);
        unsigned short s0 = (unsigned short)u, s1 = (unsigned short)(u >> 16);
        __bf16 h0, h1; __builtin_memcpy(&h0, &s0, 2); __builtin_memcpy(&h1, &s1, 2);
        Bs[n2 * 2 + 0][k] = h0; Bs[n2 * 2 + 1][k] = h1;
    }
}
// B is NxK row-major (transB): rows already match Bs[n][k]
__device__ __forceinline__ void stBt(const float* Bb, int col0, int ldb, int kk,
                                     int tid, __bf16 (*Bs)[40]) {
    #pragma unroll
    for (int i = 0; i < 4; ++i) {
        int idx = i * 128 + tid;
        int n = idx >> 3, c4 = idx & 7;
        const float4 q = *(const float4*)(Bb + (size_t)(col0 + n) * ldb + kk + c4 * 4);
        union { __bf16 h[4]; uint2 u; } p;
        p.h[0] = f2bf(q.x); p.h[1] = f2bf(q.y); p.h[2] = f2bf(q.z); p.h[3] = f2bf(q.w);
        *(uint2*)&Bs[n][c4 * 4] = p.u;
    }
}
__device__ __forceinline__ void stBt(const __bf16* Bb, int col0, int ldb, int kk,
                                     int tid, __bf16 (*Bs)[40]) {
    #pragma unroll
    for (int i = 0; i < 2; ++i) {
        int idx = i * 128 + tid;
        int n = idx >> 2, c8 = idx & 3;
        *(uint4*)&Bs[n][c8 * 8] = *(const uint4*)(Bb + (size_t)(col0 + n) * ldb + kk + c8 * 8);
    }
}

// ---------------------------------------------------------------------------
// Generic WMMA GEMM, bf16 math / f32 accumulate, bf16 output.
//   C[bz](MxN) = op( A[bz](MxK) @ B[bz](KxN) + bias[c % bias_mod] + addsrc[bz] )
// TA/TB in {float, __bf16}. transB: B stored NxK (ldb = row length).
// Block 128 thr = 4 waves, tile 64x64, wave tile 16x64, K chunks of 32.
// bf16 A panels are staged by the Tensor Data Mover (TENSORcnt).
// Requires M%64==0, N%64==0, K%32==0 (holds for all call sites).
// ---------------------------------------------------------------------------
template<typename TA, typename TB, bool RELU>
__global__ __launch_bounds__(128) void gemm_wmma(
    const TA* __restrict__ A, const TB* __restrict__ B,
    const float* __restrict__ bias, const __bf16* addsrc, __bf16* C,
    int M, int N, int K, int ldb, int transB, int bias_mod,
    long long sA, long long sB, long long sC)
{
    __shared__ __bf16 As[64][40];
    __shared__ __bf16 Bs[64][40];

    const int bz = blockIdx.z;
    const TA* Ab = A + (long long)bz * sA;
    const TB* Bb = B + (long long)bz * sB;
    const int row0 = blockIdx.y * 64;
    const int col0 = blockIdx.x * 64;
    const int tid  = threadIdx.x;
    const int wave = tid >> 5;
    const int lane = tid & 31;
    const int half = lane >> 4;
    const int l    = lane & 15;

    v8f zero = {};
    v8f acc[4];
    acc[0] = zero; acc[1] = zero; acc[2] = zero; acc[3] = zero;

    for (int kk = 0; kk < K; kk += 32) {
        stageA(Ab, row0, K, kk, tid, As);            // TDM for bf16, VALU for f32
        if (transB) stBt(Bb, col0, ldb, kk, tid, Bs);
        else        stBn(Bb, col0, ldb, kk, tid, Bs);
        waitA(Ab, tid);                              // s_wait_tensorcnt 0 (issuing wave)
        __syncthreads();

        // A fragment: two 16B chunks (k = half*8.. , 16+half*8..)
        const __bf16* ar = &As[wave * 16 + l][0];
        v16bf a = frag_cat(ar + half * 8, ar + 16 + half * 8);
        #pragma unroll
        for (int ct = 0; ct < 4; ++ct) {
            const __bf16* br = &Bs[ct * 16 + l][0];
            v16bf b = frag_cat(br + half * 8, br + 16 + half * 8);
            acc[ct] = __builtin_amdgcn_wmma_f32_16x16x32_bf16(
                false, a, false, b, (short)0, acc[ct], false, false);
        }
        __syncthreads();
    }

    #pragma unroll
    for (int ct = 0; ct < 4; ++ct) {
        #pragma unroll
        for (int vv = 0; vv < 8; ++vv) {
            int r = row0 + wave * 16 + half * 8 + vv;
            int c = col0 + ct * 16 + l;
            float val = acc[ct][vv];
            if (bias)   val += bias[c % bias_mod];
            size_t off = (size_t)((long long)bz * sC) + (size_t)r * N + c;
            if (addsrc) val += bf2f(addsrc[off]);
            if (RELU)   val = fmaxf(val, 0.0f);
            C[off] = f2bf(val);
        }
    }
}

// ---------------------------------------------------------------------------
// GRU layer: one wave per 16 sequences; gh = h @ Wh^T via WMMA (16x192, K=64).
// xp = x@Wx^T + bx precomputed (bf16, layout (seq,t,192), gates r|z|n).
// ---------------------------------------------------------------------------
__global__ __launch_bounds__(32) void gru_wmma(
    const __bf16* __restrict__ xp, const float* __restrict__ Wh,
    const float* __restrict__ bh, __bf16* __restrict__ out, int T)
{
    __shared__ __bf16 WhT[192][72];   // [n][k] = Wh[n][k] (row-major copy)
    __shared__ __bf16 hb[16][72];     // hidden state, A operand
    __shared__ float  hf[16][64];
    __shared__ float  gh[16][200];

    const int lane = threadIdx.x;
    const long long seq0 = (long long)blockIdx.x * 16;

    #pragma unroll 4
    for (int i = 0; i < 96; ++i) {                    // 3072 float4 = 192x64
        int idx = i * 32 + lane;
        int n = idx >> 4, c4 = idx & 15;
        const float4 q = *(const float4*)(Wh + (size_t)n * 64 + c4 * 4);
        union { __bf16 h[4]; uint2 u; } p;
        p.h[0] = f2bf(q.x); p.h[1] = f2bf(q.y); p.h[2] = f2bf(q.z); p.h[3] = f2bf(q.w);
        *(uint2*)&WhT[n][c4 * 4] = p.u;
    }
    for (int i = 0; i < 32; ++i) {
        int idx = i * 32 + lane; int sq = idx >> 6, u = idx & 63;
        hf[sq][u] = 0.0f; hb[sq][u] = f2bf(0.0f);
    }
    __syncthreads();

    const int half = lane >> 4, l = lane & 15;

    for (int t = 0; t < T; ++t) {
        const __bf16* hr = &hb[l][0];
        v16bf a0 = frag_cat(hr + half * 8,      hr + 16 + half * 8);
        v16bf a1 = frag_cat(hr + 32 + half * 8, hr + 48 + half * 8);
        for (int ct = 0; ct < 12; ++ct) {
            const __bf16* wr = &WhT[ct * 16 + l][0];
            v16bf b0 = frag_cat(wr + half * 8,      wr + 16 + half * 8);
            v16bf b1 = frag_cat(wr + 32 + half * 8, wr + 48 + half * 8);
            v8f acc = {};
            acc = __builtin_amdgcn_wmma_f32_16x16x32_bf16(false, a0, false, b0, (short)0, acc, false, false);
            acc = __builtin_amdgcn_wmma_f32_16x16x32_bf16(false, a1, false, b1, (short)0, acc, false, false);
            #pragma unroll
            for (int vv = 0; vv < 8; ++vv)
                gh[half * 8 + vv][ct * 16 + l] = acc[vv];
        }
        __syncthreads();

        for (int i = 0; i < 32; ++i) {
            int idx = i * 32 + lane; int sq = idx >> 6, u = idx & 63;
            long long base = ((seq0 + sq) * T + t) * 192LL;
            float xr = bf2f(xp[base + u]);
            float xz = bf2f(xp[base + 64 + u]);
            float xn = bf2f(xp[base + 128 + u]);
            float hrg = gh[sq][u] + bh[u];
            float hzg = gh[sq][64 + u] + bh[64 + u];
            float hng = gh[sq][128 + u] + bh[128 + u];
            float rg = fastsig(xr + hrg), zg = fastsig(xz + hzg);
            float cand = fasttanh(xn + rg * hng);
            float hnew = (1.0f - zg) * cand + zg * hf[sq][u];
            hf[sq][u] = hnew;
            hb[sq][u] = f2bf(hnew);
            out[((seq0 + sq) * T + t) * 64 + u] = f2bf(hnew);
        }
        __syncthreads();
    }
}

// ---------------------------------------------------------------------------
// Causal MHA: one wave per (sequence, head). T=32, DK=DV=16 (K zero-padded).
// ---------------------------------------------------------------------------
__global__ __launch_bounds__(32) void attn_wmma(
    const __bf16* __restrict__ q, const __bf16* __restrict__ k,
    const __bf16* __restrict__ v, __bf16* __restrict__ o)
{
    __shared__ __bf16 Qs[32][40];    // A: [query t][k], cols 16..31 zero
    __shared__ __bf16 Ks[32][40];    // B^T: [key t][k], cols 16..31 zero
    __shared__ __bf16 Vs[16][40];    // B^T: [d][key t]
    __shared__ float  Sf[32][33];
    __shared__ __bf16 Aw[32][40];    // A: [query t][key t]

    const int lane = threadIdx.x;
    const long long n = blockIdx.x;
    const int h = blockIdx.y;
    const long long base = n * 32LL * 64LL + (long long)h * 16;

    {
        const __bf16* qp = q + base + (long long)lane * 64;
        const __bf16* kp = k + base + (long long)lane * 64;
        const __bf16* vp = v + base + (long long)lane * 64;
        *(uint4*)&Qs[lane][0] = *(const uint4*)qp;
        *(uint4*)&Qs[lane][8] = *(const uint4*)(qp + 8);
        uint4 z; z.x = z.y = z.z = z.w = 0u;
        *(uint4*)&Qs[lane][16] = z; *(uint4*)&Qs[lane][24] = z;
        *(uint4*)&Ks[lane][0] = *(const uint4*)kp;
        *(uint4*)&Ks[lane][8] = *(const uint4*)(kp + 8);
        *(uint4*)&Ks[lane][16] = z; *(uint4*)&Ks[lane][24] = z;
        union { uint4 u[2]; __bf16 hh[16]; } vr;
        vr.u[0] = *(const uint4*)vp;
        vr.u[1] = *(const uint4*)(vp + 8);
        #pragma unroll
        for (int d = 0; d < 16; ++d) Vs[d][lane] = vr.hh[d];
    }
    __syncthreads();

    const int half = lane >> 4, l = lane & 15;

    // S = (Q K^T) / sqrt(16)
    #pragma unroll
    for (int mi = 0; mi < 2; ++mi) {
        const __bf16* qr = &Qs[mi * 16 + l][0];
        v16bf a = frag_cat(qr + half * 8, qr + 16 + half * 8);
        #pragma unroll
        for (int nj = 0; nj < 2; ++nj) {
            const __bf16* kr = &Ks[nj * 16 + l][0];
            v16bf b = frag_cat(kr + half * 8, kr + 16 + half * 8);
            v8f acc = {};
            acc = __builtin_amdgcn_wmma_f32_16x16x32_bf16(false, a, false, b, (short)0, acc, false, false);
            #pragma unroll
            for (int vv = 0; vv < 8; ++vv)
                Sf[mi * 16 + half * 8 + vv][nj * 16 + l] = acc[vv] * 0.25f;
        }
    }
    __syncthreads();

    // causal softmax: lane m owns query row m (keys j<=m valid)
    {
        const int m = lane;
        float mx = -3.0e38f;
        for (int j = 0; j <= m; ++j) mx = fmaxf(mx, Sf[m][j]);
        float s = 0.0f;
        for (int j = 0; j < 32; ++j) {
            float e = (j <= m) ? __expf(Sf[m][j] - mx) : 0.0f;
            Sf[m][j] = e; s += e;
        }
        float rs = __builtin_amdgcn_rcpf(s);
        for (int j = 0; j < 32; ++j) Aw[m][j] = f2bf(Sf[m][j] * rs);
    }
    __syncthreads();

    // O = attn @ V
    #pragma unroll
    for (int mi = 0; mi < 2; ++mi) {
        const __bf16* arow = &Aw[mi * 16 + l][0];
        v16bf a = frag_cat(arow + half * 8, arow + 16 + half * 8);
        const __bf16* vrow = &Vs[l][0];
        v16bf b = frag_cat(vrow + half * 8, vrow + 16 + half * 8);
        v8f acc = {};
        acc = __builtin_amdgcn_wmma_f32_16x16x32_bf16(false, a, false, b, (short)0, acc, false, false);
        #pragma unroll
        for (int vv = 0; vv < 8; ++vv)
            o[base + (long long)(mi * 16 + half * 8 + vv) * 64 + l] = f2bf(acc[vv]);
    }
}

// ---------------------------------------------------------------------------
// LayerNorm over last dim = 64; one wave per row. bf16 in / bf16 out.
// ---------------------------------------------------------------------------
__global__ __launch_bounds__(256) void ln64_kernel(
    const __bf16* __restrict__ x, const float* __restrict__ g,
    const float* __restrict__ b, __bf16* __restrict__ y, int rows)
{
    int wid  = (int)((blockIdx.x * (unsigned)blockDim.x + threadIdx.x) >> 5);
    int lane = threadIdx.x & 31;
    if (wid >= rows) return;
    const __bf16* xr = x + (size_t)wid * 64;
    float a0 = bf2f(xr[lane]), a1 = bf2f(xr[lane + 32]);
    float s = a0 + a1, sq = a0 * a0 + a1 * a1;
    #pragma unroll
    for (int off = 16; off > 0; off >>= 1) {
        s  += __shfl_xor(s, off);
        sq += __shfl_xor(sq, off);
    }
    float mean = s * (1.0f / 64.0f);
    float var  = sq * (1.0f / 64.0f) - mean * mean;
    float inv  = rsqrtf(var + 1e-6f);
    y[(size_t)wid * 64 + lane]      = f2bf((a0 - mean) * inv * g[lane]      + b[lane]);
    y[(size_t)wid * 64 + lane + 32] = f2bf((a1 - mean) * inv * g[lane + 32] + b[lane + 32]);
}

// ---------------------------------------------------------------------------
// Final projection (rows x 128) @ Wp^T(128x2) -> f32 (rows x 2). N=2 -> VALU.
// ---------------------------------------------------------------------------
__global__ __launch_bounds__(256) void proj2_kernel(
    const __bf16* __restrict__ hg, const float* __restrict__ Wp,
    float* __restrict__ out, int rows)
{
    int i = blockIdx.x * blockDim.x + threadIdx.x;
    if (i >= rows * 2) return;
    int r = i >> 1, c = i & 1;
    const __bf16* hr = hg + (size_t)r * 128;
    const float*  wr = Wp + (size_t)c * 128;
    float s = 0.0f;
    #pragma unroll 8
    for (int kq = 0; kq < 128; ++kq) s += bf2f(hr[kq]) * wr[kq];
    out[i] = s;
}

// ---------------------------------------------------------------------------
// f32 -> bf16 convert (vectorized), for adj pre-conversion.
// ---------------------------------------------------------------------------
__global__ __launch_bounds__(256) void cvt_bf16_kernel(
    const float* __restrict__ in, __bf16* __restrict__ out, long long n4)
{
    long long i = (long long)blockIdx.x * blockDim.x + threadIdx.x;
    if (i >= n4) return;
    const float4 q = *(const float4*)(in + i * 4);
    union { __bf16 h[4]; uint2 u; } p;
    p.h[0] = f2bf(q.x); p.h[1] = f2bf(q.y); p.h[2] = f2bf(q.z); p.h[3] = f2bf(q.w);
    *(uint2*)(out + i * 4) = p.u;
}

// ---------------------------------------------------------------------------
// Host-side orchestration
// ---------------------------------------------------------------------------
extern "C" void kernel_launch(void* const* d_in, const int* in_sizes, int n_in,
                              void* d_out, int out_size, void* d_ws, size_t ws_size,
                              hipStream_t stream)
{
    (void)in_sizes; (void)n_in; (void)out_size; (void)ws_size;

    const float* src = (const float*)d_in[0];   // (8,1024,32,64)
    const float* Hm  = (const float*)d_in[1];   // (1024,128)
    const float* adj = (const float*)d_in[2];   // (1024,1024)
    const float* W1  = (const float*)d_in[4];   // (64,128)
    const float* b1  = (const float*)d_in[5];
    const float* W2  = (const float*)d_in[6];   // (128,64)
    const float* b2  = (const float*)d_in[7];
    const float* Wm  = (const float*)d_in[8];   // (1024,128)
    const float* bm  = (const float*)d_in[9];
    const float* Wl  = (const float*)d_in[10];  // (64,64)
    const float* bl  = (const float*)d_in[11];
    const float* Wx0 = (const float*)d_in[12];  // (192,64)
    const float* Wh0 = (const float*)d_in[13];
    const float* bx0 = (const float*)d_in[14];
    const float* bh0 = (const float*)d_in[15];
    const float* Wx1 = (const float*)d_in[16];
    const float* Wh1 = (const float*)d_in[17];
    const float* bx1 = (const float*)d_in[18];
    const float* bh1 = (const float*)d_in[19];
    const float* Wq  = (const float*)d_in[20];
    const float* Wk  = (const float*)d_in[21];
    const float* Wv  = (const float*)d_in[22];
    const float* Wo  = (const float*)d_in[23];
    const float* lng = (const float*)d_in[24];
    const float* lnb = (const float*)d_in[25];
    const float* Wg1 = (const float*)d_in[26];  // (2048,128)
    const float* Wg2 = (const float*)d_in[27];  // (128,128)
    const float* Wp  = (const float*)d_in[28];  // (2,128)
    float* out = (float*)d_out;

    // bf16 workspace (with reuse)
    __bf16* Wb = (__bf16*)d_ws;
    size_t o = 0;
    __bf16* adjb = Wb + o; o += 1024ull * 1024;
    __bf16* Hnb  = Wb + o; o += 1024ull * 1024;
    __bf16* P1b  = Wb + o; o += 262144ull * 128;   // src@W1; later q; later G1
    __bf16* hAb  = Wb + o; o += 262144ull * 128;   // relu branch; later k; later hg1
    __bf16* Qbb  = Wb + o; o += 262144ull * 64;    // @W2; later v; later G2
    __bf16* zbb  = Wb + o; o += 262144ull * 64;    // z; later attn out; later hg2
    __bf16* xbb  = Wb + o; o += 262144ull * 64;    // x; later GRU2 out (r)
    __bf16* xpbb = Wb + o; o += 262144ull * 192;   // GRU input proj; later pre-LN
    __bf16* r1b  = Wb + o; o += 262144ull * 64;    // GRU1 out; later enc

    dim3 blk(128);
    const long long S46 = 1024LL * 4096, S26 = 1024LL * 2048, S17 = 1024LL * 128;

    // 0. adj -> bf16 (read 4x as TDM A operand)
    cvt_bf16_kernel<<<dim3(1024), dim3(256), 0, stream>>>(adj, adjb, 262144);
    // 1. Hn = H @ Wm^T + bm
    gemm_wmma<float, float, false><<<dim3(16, 16, 1), blk, 0, stream>>>(
        Hm, Wm, bm, nullptr, Hnb, 1024, 1024, 128, 128, 1, 1024, 0, 0, 0);
    // 2. P1 = src @ W1
    gemm_wmma<float, float, false><<<dim3(2, 4096, 1), blk, 0, stream>>>(
        src, W1, nullptr, nullptr, P1b, 262144, 128, 64, 128, 0, 1, 0, 0, 0);
    // 3. hA = relu(adj (x) P1 + b1)   batched M=1024,K=1024,N=4096
    gemm_wmma<__bf16, __bf16, true><<<dim3(64, 16, 8), blk, 0, stream>>>(
        adjb, P1b, b1, nullptr, hAb, 1024, 4096, 1024, 4096, 0, 128, 0, S46, S46);
    // 4. Qb = hA @ W2
    gemm_wmma<__bf16, float, false><<<dim3(1, 4096, 1), blk, 0, stream>>>(
        hAb, W2, nullptr, nullptr, Qbb, 262144, 64, 128, 64, 0, 1, 0, 0, 0);
    // 5. zb = adj (x) Qb + b2
    gemm_wmma<__bf16, __bf16, false><<<dim3(32, 16, 8), blk, 0, stream>>>(
        adjb, Qbb, b2, nullptr, zbb, 1024, 2048, 1024, 2048, 0, 64, 0, S26, S26);
    // 6. hA = relu(Hn (x) P1 + b1)
    gemm_wmma<__bf16, __bf16, true><<<dim3(64, 16, 8), blk, 0, stream>>>(
        Hnb, P1b, b1, nullptr, hAb, 1024, 4096, 1024, 4096, 0, 128, 0, S46, S46);
    // 7. Qb = hA @ W2
    gemm_wmma<__bf16, float, false><<<dim3(1, 4096, 1), blk, 0, stream>>>(
        hAb, W2, nullptr, nullptr, Qbb, 262144, 64, 128, 64, 0, 1, 0, 0, 0);
    // 8. zb += Hn (x) Qb + b2
    gemm_wmma<__bf16, __bf16, false><<<dim3(32, 16, 8), blk, 0, stream>>>(
        Hnb, Qbb, b2, zbb, zbb, 1024, 2048, 1024, 2048, 0, 64, 0, S26, S26);
    // 9. x = z @ Wl^T + bl
    gemm_wmma<__bf16, float, false><<<dim3(1, 4096, 1), blk, 0, stream>>>(
        zbb, Wl, bl, nullptr, xbb, 262144, 64, 64, 64, 1, 64, 0, 0, 0);
    // 10. xp = x @ Wx0^T + bx0
    gemm_wmma<__bf16, float, false><<<dim3(3, 4096, 1), blk, 0, stream>>>(
        xbb, Wx0, bx0, nullptr, xpbb, 262144, 192, 64, 64, 1, 192, 0, 0, 0);
    // 11. r1 = GRU layer 0
    gru_wmma<<<dim3(512), dim3(32), 0, stream>>>(xpbb, Wh0, bh0, r1b, 32);
    // 12. xp = r1 @ Wx1^T + bx1
    gemm_wmma<__bf16, float, false><<<dim3(3, 4096, 1), blk, 0, stream>>>(
        r1b, Wx1, bx1, nullptr, xpbb, 262144, 192, 64, 64, 1, 192, 0, 0, 0);
    // 13. r = GRU layer 1 -> xbb
    gru_wmma<<<dim3(512), dim3(32), 0, stream>>>(xpbb, Wh1, bh1, xbb, 32);
    // 14. q / k / v projections
    gemm_wmma<__bf16, float, false><<<dim3(1, 4096, 1), blk, 0, stream>>>(
        xbb, Wq, nullptr, nullptr, P1b, 262144, 64, 64, 64, 1, 1, 0, 0, 0);
    gemm_wmma<__bf16, float, false><<<dim3(1, 4096, 1), blk, 0, stream>>>(
        xbb, Wk, nullptr, nullptr, hAb, 262144, 64, 64, 64, 1, 1, 0, 0, 0);
    gemm_wmma<__bf16, float, false><<<dim3(1, 4096, 1), blk, 0, stream>>>(
        xbb, Wv, nullptr, nullptr, Qbb, 262144, 64, 64, 64, 1, 1, 0, 0, 0);
    // 15. attention -> zbb
    attn_wmma<<<dim3(8192, 4, 1), dim3(32), 0, stream>>>(P1b, hAb, Qbb, zbb);
    // 16. pre-LN = o @ Wo^T + r
    gemm_wmma<__bf16, float, false><<<dim3(1, 4096, 1), blk, 0, stream>>>(
        zbb, Wo, nullptr, xbb, xpbb, 262144, 64, 64, 64, 1, 1, 0, 0, 0);
    // 17. LayerNorm -> r1b (= enc, (8192 x 2048) contiguous)
    ln64_kernel<<<dim3(262144 / 8), dim3(256), 0, stream>>>(xpbb, lng, lnb, r1b, 262144);
    // 18. G1 = enc @ Wg1
    gemm_wmma<__bf16, float, false><<<dim3(2, 128, 1), blk, 0, stream>>>(
        r1b, Wg1, nullptr, nullptr, P1b, 8192, 128, 2048, 128, 0, 1, 0, 0, 0);
    // 19. hg1 = relu(adj (x) G1)
    gemm_wmma<__bf16, __bf16, true><<<dim3(2, 16, 8), blk, 0, stream>>>(
        adjb, P1b, nullptr, nullptr, hAb, 1024, 128, 1024, 128, 0, 1, 0, S17, S17);
    // 20. G2 = hg1 @ Wg2
    gemm_wmma<__bf16, float, false><<<dim3(2, 128, 1), blk, 0, stream>>>(
        hAb, Wg2, nullptr, nullptr, Qbb, 8192, 128, 128, 128, 0, 1, 0, 0, 0);
    // 21. hg2 = adj (x) G2
    gemm_wmma<__bf16, __bf16, false><<<dim3(2, 16, 8), blk, 0, stream>>>(
        adjb, Qbb, nullptr, nullptr, zbb, 1024, 128, 1024, 128, 0, 1, 0, S17, S17);
    // 22. out = hg2 @ Wp^T
    proj2_kernel<<<dim3(64), dim3(256), 0, stream>>>(zbb, Wp, out, 8192);
}
